// PDGCN_3736621547610
// MI455X (gfx1250) — matrix operand order repaired and checked by reading, not compiled, not run on hardware
//
#include <hip/hip_runtime.h>

typedef __attribute__((ext_vector_type(16))) _Float16 v16h;
typedef __attribute__((ext_vector_type(8)))  _Float16 v8h;
typedef __attribute__((ext_vector_type(8)))  float    v8f;

#define N_NODES 1024
#define B_BATCH 64
#define C_IN    32
#define C_OUT   32
#define K_CHEB  5
#define D_EMB   10
#define LDA     40   // LDS row stride in halves (80 bytes, multiple of 16B)
#define LDAF    36   // LDS row stride in floats for async-staged f32 A tile (144B, 16B-aligned)

__device__ __forceinline__ v16h make_frag(const _Float16* lo, const _Float16* hi) {
  v8h a = *(const v8h*)lo;
  v8h b = *(const v8h*)hi;
  v16h r;
#pragma unroll
  for (int i = 0; i < 8; ++i) { r[i] = a[i]; r[i + 8] = b[i]; }
  return r;
}

// Build a 16x32 f16 A-fragment from two 8-float runs staged as f32 in LDS.
__device__ __forceinline__ v16h cvt_frag(const float* p0, const float* p1) {
  float4 a0 = *(const float4*)p0;
  float4 a1 = *(const float4*)(p0 + 4);
  float4 a2 = *(const float4*)p1;
  float4 a3 = *(const float4*)(p1 + 4);
  v16h r;
  r[0]  = (_Float16)a0.x; r[1]  = (_Float16)a0.y; r[2]  = (_Float16)a0.z; r[3]  = (_Float16)a0.w;
  r[4]  = (_Float16)a1.x; r[5]  = (_Float16)a1.y; r[6]  = (_Float16)a1.z; r[7]  = (_Float16)a1.w;
  r[8]  = (_Float16)a2.x; r[9]  = (_Float16)a2.y; r[10] = (_Float16)a2.z; r[11] = (_Float16)a2.w;
  r[12] = (_Float16)a3.x; r[13] = (_Float16)a3.y; r[14] = (_Float16)a3.z; r[15] = (_Float16)a3.w;
  return r;
}

// Generic pointer into LDS aperture: low 32 bits carry the LDS byte address.
__device__ __forceinline__ unsigned lds_addr(const void* p) {
  return (unsigned)(unsigned long long)p;
}

// CDNA5 async copy: 16B per active lane, global -> LDS, tracked by ASYNCcnt.
__device__ __forceinline__ void async_copy_b128(unsigned lds_off, const float* gp) {
  asm volatile("global_load_async_to_lds_b128 %0, %1, off"
               :: "v"(lds_off), "v"(gp)
               : "memory");
}

// ---------------------------------------------------------------------------
// Hypernetwork: wh[n,k,i,o] = sum_d emb[n,d] * wp[d,k,i,o]   (f16 out)
// ---------------------------------------------------------------------------
__global__ __launch_bounds__(256)
void hyper_weights(const float* __restrict__ emb, const float* __restrict__ wp,
                   _Float16* __restrict__ wh) {
  int idx = blockIdx.x * 256 + threadIdx.x;      // N*K*I*O = 5,242,880
  int o   = idx & 31;
  int i   = (idx >> 5) & 31;
  int rem = idx >> 10;                            // n*K + k
  int k   = rem % K_CHEB;
  int n   = rem / K_CHEB;
  float acc = 0.f;
#pragma unroll
  for (int d = 0; d < D_EMB; ++d)
    acc += emb[n * D_EMB + d] * wp[(((size_t)d * K_CHEB + k) * C_IN + i) * C_OUT + o];
  wh[idx] = (_Float16)acc;
}

__global__ __launch_bounds__(256)
void hyper_bias(const float* __restrict__ emb, const float* __restrict__ bp,
                float* __restrict__ bias) {
  int idx = blockIdx.x * 256 + threadIdx.x;      // N*O = 32768
  int o = idx & 31;
  int n = idx >> 5;
  float acc = 0.f;
#pragma unroll
  for (int d = 0; d < D_EMB; ++d)
    acc += emb[n * D_EMB + d] * bp[d * C_OUT + o];
  bias[idx] = acc;
}

__global__ __launch_bounds__(256)
void cvt_half(const float* __restrict__ in, _Float16* __restrict__ outp) {
  int idx = blockIdx.x * 256 + threadIdx.x;      // B*N*I = 2,097,152
  outp[idx] = (_Float16)in[idx];
}

// ---------------------------------------------------------------------------
// Propagation GEMM: Y[b] (N x 32, f16) = A[b] (N x N, f32) * X[b] (N x 32, f16)
// Double-buffered LDS; A tiles stream in via GLOBAL_LOAD_ASYNC_TO_LDS_B128
// (ASYNCcnt), converted f32->f16 at fragment-build time; 8 waves issue two
// v_wmma_f32_16x16x32_f16 each per K-chunk of 32.
// ---------------------------------------------------------------------------
__global__ __launch_bounds__(256)
void prop_gemm(const float* __restrict__ A, const _Float16* __restrict__ X,
               _Float16* __restrict__ Y) {
  const int b    = blockIdx.y;
  const int m0   = blockIdx.x * 128;
  const int t    = threadIdx.x;
  const int lane = t & 31;
  const int wave = t >> 5;

  const float*    Ab = A + (size_t)b * N_NODES * N_NODES;
  const _Float16* Xb = X + (size_t)b * N_NODES * C_IN;
  _Float16*       Yb = Y + (size_t)b * N_NODES * C_IN;

  __shared__ __align__(16) float    sA[2][128 * LDAF];  // f32 tile, async-staged
  __shared__ __align__(16) _Float16 sX[2][32 * LDA];    // transposed: [col][k]

  v8f c0 = {};
  v8f c1 = {};

  const int lr   = lane & 15;
  const int ksel = lane >> 4;       // 0 or 1 (half-wave select)

  // Per-thread async slots: 1024 x 16B slots cover the 128x32 f32 tile.
  // slot = t + 256*e : row = slot/8, kp = (slot%8)*4 floats.
#define ISSUE_A(BUF, KK)                                                        \
  {                                                                             \
    _Pragma("unroll")                                                           \
    for (int e = 0; e < 4; ++e) {                                               \
      int slot = t + 256 * e;                                                   \
      int row  = slot >> 3;                                                     \
      int kp   = (slot & 7) * 4;                                                \
      async_copy_b128(lds_addr(&sA[BUF][row * LDAF + kp]),                      \
                      Ab + (size_t)(m0 + row) * N_NODES + (KK) + kp);           \
    }                                                                           \
  }

  // Prologue: start streaming the first A tile.
  ISSUE_A(0, 0)

  const int NITER = N_NODES / 32;
  for (int it = 0; it < NITER; ++it) {
    const int cur = it & 1;
    const int kk  = it * 32;

    // Stage X chunk (2 KB) transposed: sX[cur][c][k]
#pragma unroll
    for (int e = 0; e < 4; ++e) {
      int idx = t + 256 * e;
      int k   = idx >> 5;
      int cc  = idx & 31;
      sX[cur][cc * LDA + k] = Xb[(size_t)(kk + k) * C_IN + cc];
    }

    if (it + 1 < NITER) {
      ISSUE_A(cur ^ 1, kk + 32)
      // 4 copies just issued for the next buffer; in-order completion means
      // asynccnt<=4 guarantees the current buffer has fully landed.
      asm volatile("s_wait_asynccnt 0x4" ::: "memory");
    } else {
      asm volatile("s_wait_asynccnt 0x0" ::: "memory");
    }
    __syncthreads();

    // A-frag (16x32 f16): lane<16 holds K{0..7,16..23}, lane>=16 K{8..15,24..31}
    const float* arow = &sA[cur][(wave * 16 + lr) * LDAF];
    v16h af = cvt_frag(arow + ksel * 8, arow + ksel * 8 + 16);
    // B-frags (32x16 f16): lane<16 -> K{0..7,16..23}, lane>=16 -> K{8..15,24..31}, col = lr
    v16h b0 = make_frag(&sX[cur][lr * LDA + ksel * 8],
                        &sX[cur][lr * LDA + ksel * 8 + 16]);
    v16h b1 = make_frag(&sX[cur][(16 + lr) * LDA + ksel * 8],
                        &sX[cur][(16 + lr) * LDA + ksel * 8 + 16]);

    c0 = __builtin_amdgcn_wmma_f32_16x16x32_f16(false, af, false, b0, (short)0, c0, false, false);
    c1 = __builtin_amdgcn_wmma_f32_16x16x32_f16(false, af, false, b1, (short)0, c1, false, false);
    __syncthreads();   // all waves done reading before next iteration overwrites
  }
#undef ISSUE_A

  // C/D layout: VGPR r -> M = r (lanes 0..15) / r+8 (lanes 16..31); N = lr
#pragma unroll
  for (int r = 0; r < 8; ++r) {
    int row = m0 + wave * 16 + ksel * 8 + r;
    Yb[(size_t)row * C_IN + lr]      = (_Float16)c0[r];
    Yb[(size_t)row * C_IN + 16 + lr] = (_Float16)c1[r];
  }
}

// ---------------------------------------------------------------------------
// Combine: out[b,n,o] = sum_k sum_i xg_k[b,n,i] * W[n,k,i,o] + bias[n,o]
// One workgroup per node n: 64(batch) x 160(k*i) x 32(o) GEMM via WMMA.
// ---------------------------------------------------------------------------
__global__ __launch_bounds__(256)
void combine_kernel(const _Float16* __restrict__ g0, const _Float16* __restrict__ g1,
                    const _Float16* __restrict__ g2, const _Float16* __restrict__ g3,
                    const _Float16* __restrict__ g4,
                    const _Float16* __restrict__ wh, const float* __restrict__ bias,
                    float* __restrict__ out) {
  const int n    = blockIdx.x;
  const int t    = threadIdx.x;
  const int lane = t & 31;
  const int wave = t >> 5;
  const int mt   = wave & 3;     // batch tile (4 x 16 = 64)
  const int ht   = wave >> 2;    // output-col half (2 x 16 = 32)
  const int lr   = lane & 15;
  const int ksel = lane >> 4;

  const _Float16* srcs[K_CHEB] = {g0, g1, g2, g3, g4};

  __shared__ __align__(16) _Float16 sA[64 * LDA];   // [batch][i]
  __shared__ __align__(16) _Float16 sB[32 * LDA];   // transposed: [o][i]

  v8f c = {};

  for (int k = 0; k < K_CHEB; ++k) {
    const _Float16* src = srcs[k];
    {
      int row  = t >> 2;            // 0..63  (batch)
      int part = (t & 3) * 8;       // 0,8,16,24
      v8h v = *(const v8h*)(src + ((size_t)row * N_NODES + n) * C_IN + part);
      *(v8h*)(&sA[row * LDA + part]) = v;
    }
#pragma unroll
    for (int e = 0; e < 4; ++e) {
      int idx = t + 256 * e;        // 32*32 = 1024 elements
      int i   = idx >> 5;
      int o   = idx & 31;
      sB[o * LDA + i] = wh[(((size_t)n * K_CHEB + k) * C_IN + i) * C_OUT + o];
    }
    __syncthreads();

    v16h af = make_frag(&sA[(mt * 16 + lr) * LDA + ksel * 8],
                        &sA[(mt * 16 + lr) * LDA + ksel * 8 + 16]);
    v16h bf = make_frag(&sB[(ht * 16 + lr) * LDA + ksel * 8],
                        &sB[(ht * 16 + lr) * LDA + ksel * 8 + 16]);

    c = __builtin_amdgcn_wmma_f32_16x16x32_f16(false, af, false, bf, (short)0, c, false, false);
    __syncthreads();
  }

  float bv = bias[n * C_OUT + ht * 16 + lr];
#pragma unroll
  for (int r = 0; r < 8; ++r) {
    int brow = mt * 16 + ksel * 8 + r;   // batch index
    out[((size_t)brow * N_NODES + n) * C_OUT + ht * 16 + lr] = c[r] + bv;
  }
}

// ---------------------------------------------------------------------------
extern "C" void kernel_launch(void* const* d_in, const int* in_sizes, int n_in,
                              void* d_out, int out_size, void* d_ws, size_t ws_size,
                              hipStream_t stream) {
  (void)in_sizes; (void)n_in; (void)out_size; (void)ws_size;

  const float* x   = (const float*)d_in[0];   // (B, N, I)
  const float* s0  = (const float*)d_in[1];   // (B, N, N)
  const float* s1  = (const float*)d_in[2];   // (B, N, N)
  const float* emb = (const float*)d_in[3];   // (N, D)
  const float* wp  = (const float*)d_in[4];   // (D, K, I, O)
  const float* bp  = (const float*)d_in[5];   // (D, O)
  float* out = (float*)d_out;                 // (B, N, O)

  char* ws = (char*)d_ws;
  size_t off = 0;
  _Float16* wh = (_Float16*)(ws + off); off += (size_t)N_NODES * K_CHEB * C_IN * C_OUT * 2; // 10 MB
  float* bias  = (float*)(ws + off);    off += (size_t)N_NODES * C_OUT * 4;                 // 128 KB
  const size_t xbytes = (size_t)B_BATCH * N_NODES * C_IN * 2;                               // 4 MB each
  _Float16* xh  = (_Float16*)(ws + off); off += xbytes;
  _Float16* x10 = (_Float16*)(ws + off); off += xbytes;
  _Float16* x20 = (_Float16*)(ws + off); off += xbytes;
  _Float16* x11 = (_Float16*)(ws + off); off += xbytes;
  _Float16* x21 = (_Float16*)(ws + off); off += xbytes;

  // Hypernetwork + f16 conversion of x
  hipLaunchKernelGGL(hyper_weights, dim3(20480), dim3(256), 0, stream, emb, wp, wh);
  hipLaunchKernelGGL(hyper_bias,    dim3(128),   dim3(256), 0, stream, emb, bp, bias);
  hipLaunchKernelGGL(cvt_half,      dim3(8192),  dim3(256), 0, stream, x, xh);

  // Chebyshev propagation: x1 = A*x ; x2 = A*x1 for each support
  dim3 ggrid(N_NODES / 128, B_BATCH);
  hipLaunchKernelGGL(prop_gemm, ggrid, dim3(256), 0, stream, s0, xh,  x10);
  hipLaunchKernelGGL(prop_gemm, ggrid, dim3(256), 0, stream, s0, x10, x20);
  hipLaunchKernelGGL(prop_gemm, ggrid, dim3(256), 0, stream, s1, xh,  x11);
  hipLaunchKernelGGL(prop_gemm, ggrid, dim3(256), 0, stream, s1, x11, x21);

  // Node-specific grouped GEMM + bias
  hipLaunchKernelGGL(combine_kernel, dim3(N_NODES), dim3(256), 0, stream,
                     xh, x10, x20, x11, x21, wh, bias, out);
}
